// ConvNet_37649683317479
// MI455X (gfx1250) — matrix operand order repaired and checked by reading, not compiled, not run on hardware
//
#include <hip/hip_runtime.h>
#include <hip/hip_bf16.h>

#define NN 50000
#define NE 800000
#define UNITS 96
#define HALFU 48
#define KNN_DIM 16
#define NL 3
#define EPSV 1e-5f

typedef __attribute__((ext_vector_type(16))) __bf16 v16bf;
typedef __attribute__((ext_vector_type(8)))  float  v8f;

// native hw convert (gfx1250 has BF16 VALU/cvt support; lets clang emit v_cvt_*bf16_f32)
__device__ __forceinline__ __bf16 f2bf(float f) { return (__bf16)f; }

__device__ __forceinline__ v8f vzero8() {
    v8f z = {0.f,0.f,0.f,0.f,0.f,0.f,0.f,0.f};
    return z;
}

// ---------------- trivial elementwise kernels ----------------

__global__ void k_zero(float* __restrict__ p, int n) {
    int t = blockIdx.x * blockDim.x + threadIdx.x;
    if (t < n) p[t] = 0.f;
}

// Wt[l][n][k] = bf16(W[l][k][n])
__global__ void k_tposew(const float* __restrict__ W, __bf16* __restrict__ Wt,
                         int layers, int K, int N) {
    int t = blockIdx.x * blockDim.x + threadIdx.x;
    int tot = layers * K * N;
    if (t >= tot) return;
    int l = t / (K * N);
    int r = t % (K * N);
    int n = r / K;
    int k = r % K;
    Wt[t] = f2bf(W[l * K * N + k * N + n]);
}

__global__ void k_node_embed(const float* __restrict__ pos, const float* __restrict__ W,
                             const float* __restrict__ b, float* __restrict__ x) {
    int t = blockIdx.x * blockDim.x + threadIdx.x;
    if (t >= NN * UNITS) return;
    int n = t / UNITS, u = t % UNITS;
    x[t] = pos[n * 2] * W[u] + pos[n * 2 + 1] * W[UNITS + u] + b[u];
}

__global__ void k_edge_embed(const float* __restrict__ knn, const float* __restrict__ dist,
                             const float* __restrict__ dW, const float* __restrict__ db,
                             const float* __restrict__ kW, float* __restrict__ e) {
    int t = blockIdx.x * blockDim.x + threadIdx.x;
    if (t >= NE * UNITS) return;
    int ed = t / UNITS, u = t % UNITS;
    float v;
    if (u < HALFU) {
        v = dist[ed] * dW[u] + db[u];
    } else {
        int c = u - HALFU;
        const float* kr = knn + (size_t)ed * KNN_DIM;
        float s = 0.f;
#pragma unroll
        for (int k = 0; k < KNN_DIM; ++k) s += kr[k] * kW[k * HALFU + c];
        v = s;
    }
    e[t] = v;
}

// msg = relu(x[src]+e); agg[dst] += msg   (skip zeros -> fewer atomics)
__global__ void k_msg_agg(const float* __restrict__ x, const float* __restrict__ e,
                          const int* __restrict__ src, const int* __restrict__ dst,
                          float* __restrict__ agg) {
    int t = blockIdx.x * blockDim.x + threadIdx.x;
    if (t >= NE * UNITS) return;
    int ed = t / UNITS, u = t % UNITS;
    int s = src[ed], d = dst[ed];
    float m = x[s * UNITS + u] + e[t];
    if (m > 0.f) atomicAdd(&agg[d * UNITS + u], m);
}

// ---------------- WMMA kernels ----------------
// A-frag (16x32 bf16): lane m=lane&15 is row m; element e -> K = half*16 + e (contiguous, same order as B)
// B-frag: lane holds column n=lane&15; element e -> K = half*16 + e (contiguous in transposed weights)
// C/D:   VGPR r, lanes 0-15 -> M=r, lanes 16-31 -> M=r+8; N = lane&15 (+16*j tile)

// x = x + relu(((1+eps)x + agg) @ convW + convB)
__global__ void __launch_bounds__(256) k_node_conv(const float* __restrict__ agg,
                                                   const __bf16* __restrict__ Wt,  // [96][96]
                                                   const float* __restrict__ bias,
                                                   float* __restrict__ x) {
    __shared__ __align__(32) __bf16 wl[UNITS * UNITS];
    for (int i = threadIdx.x; i < UNITS * UNITS / 8; i += blockDim.x)
        ((uint4*)wl)[i] = ((const uint4*)Wt)[i];
    __syncthreads();

    int wave = threadIdx.x >> 5, lane = threadIdx.x & 31;
    int tile = blockIdx.x * 8 + wave;
    if (tile >= NN / 16) return;
    int m = lane & 15, half = lane >> 4;

    const float* xr = x   + (size_t)(tile * 16 + m) * UNITS;
    const float* ar = agg + (size_t)(tile * 16 + m) * UNITS;

    v8f acc[6];
#pragma unroll
    for (int j = 0; j < 6; ++j) acc[j] = vzero8();

#pragma unroll
    for (int c = 0; c < 3; ++c) {
        int kk = c * 32;
        v16bf a;
#pragma unroll
        for (int i = 0; i < 8; ++i) {
            int k0 = kk + half * 16 + i;
            int k1 = kk + half * 16 + 8 + i;
            a[i]     = f2bf((1.0f + EPSV) * xr[k0] + ar[k0]);
            a[8 + i] = f2bf((1.0f + EPSV) * xr[k1] + ar[k1]);
        }
#pragma unroll
        for (int j = 0; j < 6; ++j) {
            int n = j * 16 + m;
            v16bf b = *(const v16bf*)&wl[n * UNITS + kk + half * 16];
            acc[j] = __builtin_amdgcn_wmma_f32_16x16x32_bf16(
                false, a, false, b, (short)0, acc[j], false, false);
        }
    }
#pragma unroll
    for (int j = 0; j < 6; ++j) {
        int col = j * 16 + m;
        float bv = bias[col];
#pragma unroll
        for (int r = 0; r < 8; ++r) {
            int R = tile * 16 + half * 8 + r;
            float v = acc[j][r] + bv;
            v = v > 0.f ? v : 0.f;
            x[(size_t)R * UNITS + col] += v;
        }
    }
}

// e = e + relu(concat(x[src], x[dst], e) @ elinW + elinB)   (K = 288)
__global__ void __launch_bounds__(256) k_edge_lin(const float* __restrict__ x,
                                                  float* __restrict__ e,
                                                  const int* __restrict__ src,
                                                  const int* __restrict__ dst,
                                                  const __bf16* __restrict__ Wt,  // [96][288]
                                                  const float* __restrict__ bias) {
    __shared__ __align__(32) __bf16 wl[UNITS * 3 * UNITS];
    for (int i = threadIdx.x; i < UNITS * 3 * UNITS / 8; i += blockDim.x)
        ((uint4*)wl)[i] = ((const uint4*)Wt)[i];
    __syncthreads();

    int wave = threadIdx.x >> 5, lane = threadIdx.x & 31;
    int tile = blockIdx.x * 8 + wave;                // 50000 tiles exactly
    int m = lane & 15, half = lane >> 4;
    int et = tile * 16 + m;
    int s = src[et], d = dst[et];
    const float* px = x + (size_t)s  * UNITS;
    const float* pd = x + (size_t)d  * UNITS;
    const float* pe = e + (size_t)et * UNITS;

    v8f acc[6];
#pragma unroll
    for (int j = 0; j < 6; ++j) acc[j] = vzero8();

#pragma unroll
    for (int c = 0; c < 9; ++c) {
        int kk = c * 32;
        const float* rp = (kk < 96) ? (px + kk) : (kk < 192) ? (pd + kk - 96) : (pe + kk - 192);
        v16bf a;
#pragma unroll
        for (int i = 0; i < 8; ++i) {
            a[i]     = f2bf(rp[half * 16 + i]);
            a[8 + i] = f2bf(rp[half * 16 + 8 + i]);
        }
#pragma unroll
        for (int j = 0; j < 6; ++j) {
            int n = j * 16 + m;
            v16bf b = *(const v16bf*)&wl[n * (3 * UNITS) + kk + half * 16];
            acc[j] = __builtin_amdgcn_wmma_f32_16x16x32_bf16(
                false, a, false, b, (short)0, acc[j], false, false);
        }
    }
#pragma unroll
    for (int j = 0; j < 6; ++j) {
        int col = j * 16 + m;
        float bv = bias[col];
#pragma unroll
        for (int r = 0; r < 8; ++r) {
            int Ro = tile * 16 + half * 8 + r;
            float v = acc[j][r] + bv;
            v = v > 0.f ? v : 0.f;
            e[(size_t)Ro * UNITS + col] += v;
        }
    }
}

// out = prelu(e @ W1 + b1) @ W2 + b2
__global__ void __launch_bounds__(256) k_head(const float* __restrict__ e,
                                              const __bf16* __restrict__ W1t,  // [48][96]
                                              const float* __restrict__ b1,
                                              const float* __restrict__ prelu,
                                              const float* __restrict__ W2,   // [48]
                                              const float* __restrict__ b2,
                                              float* __restrict__ out) {
    __shared__ __align__(32) __bf16 wl[HALFU * UNITS];
    __shared__ float zbuf[8][16][HALFU];
    for (int i = threadIdx.x; i < HALFU * UNITS / 8; i += blockDim.x)
        ((uint4*)wl)[i] = ((const uint4*)W1t)[i];
    __syncthreads();

    int wave = threadIdx.x >> 5, lane = threadIdx.x & 31;
    int tile = blockIdx.x * 8 + wave;                // 50000 tiles exactly
    int m = lane & 15, half = lane >> 4;
    const float* pe = e + (size_t)(tile * 16 + m) * UNITS;
    float pa = prelu[0];

    v8f acc[3];
#pragma unroll
    for (int j = 0; j < 3; ++j) acc[j] = vzero8();

#pragma unroll
    for (int c = 0; c < 3; ++c) {
        int kk = c * 32;
        v16bf a;
#pragma unroll
        for (int i = 0; i < 8; ++i) {
            a[i]     = f2bf(pe[kk + half * 16 + i]);
            a[8 + i] = f2bf(pe[kk + half * 16 + 8 + i]);
        }
#pragma unroll
        for (int j = 0; j < 3; ++j) {
            int n = j * 16 + m;
            v16bf b = *(const v16bf*)&wl[n * UNITS + kk + half * 16];
            acc[j] = __builtin_amdgcn_wmma_f32_16x16x32_bf16(
                false, a, false, b, (short)0, acc[j], false, false);
        }
    }
#pragma unroll
    for (int j = 0; j < 3; ++j) {
        int col = j * 16 + m;
        float bb = b1[col];
#pragma unroll
        for (int r = 0; r < 8; ++r) {
            float z = acc[j][r] + bb;
            z = z > 0.f ? z : pa * z;
            zbuf[wave][half * 8 + r][col] = z;
        }
    }
    // intra-wave LDS RAW: wait on DScnt (wave32 lockstep => all stores issued)
    asm volatile("s_wait_dscnt 0" ::: "memory");
    if (half == 0) {
        float s2 = 0.f;
#pragma unroll
        for (int c2 = 0; c2 < HALFU; ++c2) s2 += zbuf[wave][m][c2] * W2[c2];
        out[tile * 16 + m] = s2 + b2[0];
    }
}

// ---------------- launch ----------------

extern "C" void kernel_launch(void* const* d_in, const int* in_sizes, int n_in,
                              void* d_out, int out_size, void* d_ws, size_t ws_size,
                              hipStream_t stream) {
    (void)in_sizes; (void)n_in; (void)out_size; (void)ws_size;
    const float* pos     = (const float*)d_in[0];
    const float* eknn    = (const float*)d_in[1];
    const float* edist   = (const float*)d_in[2];
    const int*   eidx    = (const int*)  d_in[3];
    const float* node_W  = (const float*)d_in[4];
    const float* node_b  = (const float*)d_in[5];
    const float* dist_W  = (const float*)d_in[6];
    const float* dist_b  = (const float*)d_in[7];
    const float* knn_W   = (const float*)d_in[8];
    const float* conv_W  = (const float*)d_in[9];
    const float* conv_b  = (const float*)d_in[10];
    const float* elin_W  = (const float*)d_in[11];
    const float* elin_b  = (const float*)d_in[12];
    const float* mlp_W1  = (const float*)d_in[13];
    const float* mlp_b1  = (const float*)d_in[14];
    const float* prelu_a = (const float*)d_in[15];
    const float* mlp_W2  = (const float*)d_in[16];
    const float* mlp_b2  = (const float*)d_in[17];
    const int* src = eidx;
    const int* dst = eidx + NE;

    char* ws = (char*)d_ws;
    size_t off = 0;
    auto take = [&](size_t bytes) -> char* {
        off = (off + 255) & ~(size_t)255;
        char* p = ws + off;
        off += bytes;
        return p;
    };
    float*  x      = (float*)take((size_t)NN * UNITS * 4);
    float*  agg    = (float*)take((size_t)NN * UNITS * 4);
    float*  ebuf   = (float*)take((size_t)NE * UNITS * 4);
    __bf16* convWt = (__bf16*)take((size_t)NL * UNITS * UNITS * 2);
    __bf16* elinWt = (__bf16*)take((size_t)(NL + 1) * UNITS * 3 * UNITS * 2);
    __bf16* mlpW1t = (__bf16*)take((size_t)HALFU * UNITS * 2);

    const int B = 256;
    // one-time weight transposes (f32 -> bf16, [K][N] -> [N][K])
    k_tposew<<<(NL * UNITS * UNITS + B - 1) / B, B, 0, stream>>>(conv_W, convWt, NL, UNITS, UNITS);
    k_tposew<<<((NL + 1) * 3 * UNITS * UNITS + B - 1) / B, B, 0, stream>>>(elin_W, elinWt, NL + 1, 3 * UNITS, UNITS);
    k_tposew<<<(UNITS * HALFU + B - 1) / B, B, 0, stream>>>(mlp_W1, mlpW1t, 1, UNITS, HALFU);

    k_node_embed<<<(NN * UNITS + B - 1) / B, B, 0, stream>>>(pos, node_W, node_b, x);
    k_edge_embed<<<(NE * UNITS + B - 1) / B, B, 0, stream>>>(eknn, edist, dist_W, dist_b, knn_W, ebuf);

    const int nodeTiles = NN / 16;            // 3125
    const int edgeBlocks = NE / 16 / 8;       // 6250 blocks of 8 wave-tiles
    for (int l = 0; l < NL; ++l) {
        k_zero<<<(NN * UNITS + B - 1) / B, B, 0, stream>>>(agg, NN * UNITS);
        k_msg_agg<<<(NE * UNITS + B - 1) / B, B, 0, stream>>>(x, ebuf, src, dst, agg);
        k_node_conv<<<(nodeTiles + 7) / 8, B, 0, stream>>>(
            agg, convWt + (size_t)l * UNITS * UNITS, conv_b + l * UNITS, x);
        k_edge_lin<<<edgeBlocks, B, 0, stream>>>(
            x, ebuf, src, dst, elinWt + (size_t)l * UNITS * 3 * UNITS, elin_b + l * UNITS);
    }
    k_edge_lin<<<edgeBlocks, B, 0, stream>>>(
        x, ebuf, src, dst, elinWt + (size_t)NL * UNITS * 3 * UNITS, elin_b + NL * UNITS);
    k_head<<<edgeBlocks, B, 0, stream>>>(ebuf, mlpW1t, mlp_b1, prelu_a, mlp_W2, mlp_b2,
                                         (float*)d_out);
}
